// SelfAttentionEsDot_11304353923148
// MI455X (gfx1250) — compile-verified, matrix operand order checked by reading, not run on hardware
//
#include <hip/hip_runtime.h>

typedef __attribute__((ext_vector_type(16))) _Float16 v16h;
typedef __attribute__((ext_vector_type(8)))  _Float16 v8h;
typedef __attribute__((ext_vector_type(8)))  float    v8f;
typedef __attribute__((ext_vector_type(4)))  float    v4f;

#define BATCH 2
#define CH    8
#define HH    128
#define WW    128
#define DK    96            // padded K: d = 8*3*3 = 72 -> 96
#define WN    7
#define PAD   3
#define PH    (HH + 2*PAD)  // 134 (padded rows)
#define PW    (WW + 2*PAD)  // 134 (padded cols)
#define NPAD  (BATCH * PH * PW)   // 35912 padded pixels

// ---------------- kernel 1: 1x1 convs on padded grid -----------------------
// interior: phi0 = Wphi*pan (f16, unpadded layout), g = Wg*u (f32, padded)
// halo:     g = 0
__global__ void conv1x1_kernel(const float* __restrict__ pan,
                               const float* __restrict__ u,
                               const float* __restrict__ Wphi,
                               const float* __restrict__ Wg,
                               _Float16* __restrict__ phi0,  // (b,h,w,8) f16, unpadded
                               float* __restrict__ gbuf)     // (b,PH,PW,8) f32, padded
{
    int t = blockIdx.x * blockDim.x + threadIdx.x;
    if (t >= NPAD) return;
    int px = t % PW;
    int py = (t / PW) % PH;
    int b  = t / (PH * PW);
    int x = px - PAD, y = py - PAD;

    float* gdst = gbuf + (size_t)t * CH;
    if (x < 0 || x >= WW || y < 0 || y >= HH) {
        v4f z = {};
        *(v4f*)gdst       = z;
        *(v4f*)(gdst + 4) = z;
        return;
    }

    float pv[CH], uv[CH];
#pragma unroll
    for (int c = 0; c < CH; ++c) {
        size_t idx = (((size_t)b * CH + c) * HH + y) * WW + x;
        pv[c] = pan[idx];
        uv[c] = u[idx];
    }

    v8h ph = {};
    v4f g0 = {}, g1 = {};
#pragma unroll
    for (int o = 0; o < CH; ++o) {
        float ap = 0.f, ag = 0.f;
#pragma unroll
        for (int i = 0; i < CH; ++i) {
            ap = fmaf(Wphi[o * CH + i], pv[i], ap);
            ag = fmaf(Wg  [o * CH + i], uv[i], ag);
        }
        ph[o] = (_Float16)ap;
        if (o < 4) g0[o] = ag; else g1[o - 4] = ag;
    }
    *(v8h*)(phi0 + ((((size_t)b * HH + y) * WW + x) * CH)) = ph;
    *(v4f*)gdst       = g0;
    *(v4f*)(gdst + 4) = g1;
}

// ---------------- kernel 2: 72-dim theta descriptors on padded grid --------
// torch unfold order ch*9 + i*3 + j; halo pixels get all-zero descriptors
__global__ void theta_kernel(const _Float16* __restrict__ phi0,
                             _Float16* __restrict__ theta)   // (b,PH,PW,96) f16
{
    int t = blockIdx.x * blockDim.x + threadIdx.x;
    if (t >= NPAD) return;
    int px = t % PW;
    int py = (t / PW) % PH;
    int b  = t / (PH * PW);
    int x = px - PAD, y = py - PAD;
    bool interior = (x >= 0) && (x < WW) && (y >= 0) && (y < HH);

    alignas(16) _Float16 tl[DK];
#pragma unroll
    for (int k = 0; k < DK; ++k) tl[k] = (_Float16)0.f;

    if (interior) {
#pragma unroll
        for (int i = 0; i < 3; ++i) {
#pragma unroll
            for (int j = 0; j < 3; ++j) {
                int hh = y + i - 1, ww2 = x + j - 1;
                v8h nb = {};
                if (hh >= 0 && hh < HH && ww2 >= 0 && ww2 < WW)
                    nb = *(const v8h*)(phi0 + (((size_t)b * HH + hh) * WW + ww2) * CH);
#pragma unroll
                for (int c = 0; c < CH; ++c) tl[c * 9 + i * 3 + j] = nb[c];
            }
        }
    }

    _Float16* dst = theta + (size_t)t * DK;
#pragma unroll
    for (int s = 0; s < DK / 8; ++s)
        *(v8h*)(dst + s * 8) = *(const v8h*)(tl + s * 8);
}

// ---------------- kernel 3: banded WMMA attention (branch-free loads) ------
// one wave32 per 16-pixel row segment; 4 waves per block
__global__ void __launch_bounds__(128)
attn_kernel(const _Float16* __restrict__ theta,
            const float* __restrict__ gbuf,
            float* __restrict__ out)
{
    __shared__ float sc[4][16][WN][32];   // per-wave score scratch (56 KB)

    const int wave = threadIdx.x >> 5;
    const int lane = threadIdx.x & 31;
    const int seg  = blockIdx.x * 4 + wave;   // 0 .. 2047
    const int wt = seg & 7;
    const int h  = (seg >> 3) & (HH - 1);
    const int b  = seg >> 10;
    const int w0 = wt << 4;

    const int M  = lane & 15;
    const int hi = lane >> 4;

    // hint: g rows for phase 2 will be needed soon
    __builtin_prefetch(gbuf + (((size_t)b * PH + (h + PAD)) * PW + (w0 + PAD)) * CH, 0, 3);

    // ---- A fragments: theta rows of the 16 pixels, K = 0..95 in 3 steps ----
    v16h A[3];
    {
        const _Float16* ar = theta
            + (((size_t)b * PH + (h + PAD)) * PW + (w0 + M + PAD)) * DK + hi * 8;
#pragma unroll
        for (int s = 0; s < 3; ++s) {
            v8h a0 = *(const v8h*)(ar + 32 * s);
            v8h a1 = *(const v8h*)(ar + 32 * s + 16);
            A[s] = __builtin_shufflevector(a0, a1, 0, 1, 2, 3, 4, 5, 6, 7,
                                                  8, 9, 10, 11, 12, 13, 14, 15);
        }
    }

    // ---- banded score GEMM: per dy, scores vs 22 (->32) neighbor columns --
    // halo descriptors are zero, so boundary scores come out 0 (matches ref)
    const _Float16* tbase = theta + ((size_t)b * PH * PW) * DK + hi * 8;
    for (int dy = 0; dy < WN; ++dy) {
        int pry = h + dy;                           // padded row = (h+dy-3)+PAD
#pragma unroll
        for (int nt = 0; nt < 2; ++nt) {
            int qx = w0 - PAD + nt * 16 + M;        // neighbor x (image coords)
            int qxc = min(qx, WW + PAD - 1);        // clamp unused cols into halo
            const _Float16* br = tbase + ((size_t)pry * PW + (qxc + PAD)) * DK;

            v8f c = {};
#pragma unroll
            for (int s = 0; s < 3; ++s) {
                v8h b0 = *(const v8h*)(br + 32 * s);
                v8h b1 = *(const v8h*)(br + 32 * s + 16);
                v16h Bf = __builtin_shufflevector(b0, b1, 0, 1, 2, 3, 4, 5, 6, 7,
                                                         8, 9, 10, 11, 12, 13, 14, 15);
                c = __builtin_amdgcn_wmma_f32_16x16x32_f16(
                        false, A[s], false, Bf, (short)0, c, false, false);
            }
#pragma unroll
            for (int e = 0; e < 8; ++e)
                sc[wave][hi * 8 + e][dy][nt * 16 + M] = c[e];
        }
    }

    __syncthreads();

    // ---- phase 2: softmax over 49 + weighted g aggregation ----------------
    // all 32 lanes: lane = (pixel m, channel-half); halo g is zero
    {
        const int m   = lane & 15;
        const int ch4 = (lane >> 4) * 4;
        const int wx  = w0 + m;

        float mx = -3.0e38f;
#pragma unroll
        for (int dy = 0; dy < WN; ++dy)
#pragma unroll
            for (int dx = 0; dx < WN; ++dx)
                mx = fmaxf(mx, sc[wave][m][dy][m + dx]);

        float den = 0.f;
        v4f acc = {};
        const float* grow = gbuf + ((size_t)b * PH * PW) * CH + ch4;
        for (int dy = 0; dy < WN; ++dy) {
            size_t rbase = (size_t)(h + dy) * PW;   // padded row
#pragma unroll
            for (int dx = 0; dx < WN; ++dx) {
                float e = __expf(sc[wave][m][dy][m + dx] - mx);
                den += e;                            // OOB q: score 0, g 0 — matches ref
                v4f gq = *(const v4f*)(grow + (rbase + (size_t)(wx + dx)) * CH);
                acc += gq * e;
            }
        }
        float inv = 1.0f / den;
#pragma unroll
        for (int c = 0; c < 4; ++c)
            out[(((size_t)b * CH + (ch4 + c)) * HH + h) * WW + wx] = acc[c] * inv;
    }
}

// ---------------------------------------------------------------------------
extern "C" void kernel_launch(void* const* d_in, const int* in_sizes, int n_in,
                              void* d_out, int out_size, void* d_ws, size_t ws_size,
                              hipStream_t stream)
{
    (void)in_sizes; (void)n_in; (void)out_size; (void)ws_size;

    const float* u    = (const float*)d_in[0];
    const float* pan  = (const float*)d_in[1];
    const float* Wphi = (const float*)d_in[2];
    const float* Wg   = (const float*)d_in[3];
    // d_in[4] = patch_size (3), d_in[5] = window_size (7): fixed by reference

    const size_t theta_bytes = (size_t)NPAD * DK * sizeof(_Float16);          // ~6.9 MB
    const size_t phi0_bytes  = (size_t)BATCH * HH * WW * CH * sizeof(_Float16); // 0.5 MB

    _Float16* theta = (_Float16*)d_ws;
    _Float16* phi0  = (_Float16*)((char*)d_ws + theta_bytes);
    float*    gbuf  = (float*)((char*)d_ws + theta_bytes + phi0_bytes);       // ~1.15 MB
    float*    out   = (float*)d_out;

    const int blocks_pad = (NPAD + 255) / 256;        // 141
    conv1x1_kernel<<<blocks_pad, 256, 0, stream>>>(pan, u, Wphi, Wg, phi0, gbuf);
    theta_kernel  <<<blocks_pad, 256, 0, stream>>>(phi0, theta);

    const int nseg = BATCH * HH * (WW / 16);          // 2048 wave-segments
    attn_kernel<<<nseg / 4, 128, 0, stream>>>(theta, gbuf, out);
}
